// PRNNCore_88596585382232
// MI455X (gfx1250) — matrix operand order repaired and checked by reading, not compiled, not run on hardware
//
#include <hip/hip_runtime.h>
#include <cstdint>
#include <cstddef>

// ---------------- types ----------------
typedef __attribute__((ext_vector_type(16))) __bf16 v16bf;
typedef __attribute__((ext_vector_type(8)))  __bf16 v8bf;
typedef __attribute__((ext_vector_type(8)))  float  v8f;

static constexpr int BATCH = 8192;
static constexpr int HID   = 400;
static constexpr int GATE  = 1600;   // 4*H
static constexpr int KL    = 896;    // padded layer K: 64(x,r,pad,d) + 416(prev_h) + 416(below)
static constexpr int KH    = 1216;   // padded hat K: 3*400 -> 1216
static constexpr int YD    = 121;
static constexpr int NP_GATE = 1664; // GATE padded to multiple of 128 rows (13*128)

__device__ __forceinline__ __bf16 f2bf(float f) {
    unsigned u = __builtin_bit_cast(unsigned, f);
    unsigned short s = (unsigned short)((u + 0x7FFFu + ((u >> 16) & 1u)) >> 16);
    return __builtin_bit_cast(__bf16, s);
}
__device__ __forceinline__ float bf2f(__bf16 h) {
    unsigned short s = __builtin_bit_cast(unsigned short, h);
    unsigned u = ((unsigned)s) << 16;
    return __builtin_bit_cast(float, u);
}
__device__ __forceinline__ float sigf(float x) { return 1.0f / (1.0f + expf(-x)); }

// 32-bit LDS byte offset of a generic pointer to a __shared__ object
__device__ __forceinline__ unsigned lds_lo(void* p) {
    return (unsigned)(size_t)(__attribute__((address_space(3))) void*)p;
}
// CDNA5 async global->LDS, 16B per lane, tracked by ASYNCcnt
__device__ __forceinline__ void async_cp16(unsigned lds, unsigned long long g) {
    asm volatile("global_load_async_to_lds_b128 %0, %1, off" :: "v"(lds), "v"(g) : "memory");
}
template <int N> __device__ __forceinline__ void wait_asynccnt() {
    asm volatile("s_wait_asynccnt %0" :: "n"(N) : "memory");
}

// ---------------- WMMA GEMM:  C[M x N] = A[M x K] * W[N x K]^T + bias ----------------
// M = gridDim.x*256 (rows always valid).  W has >= gridDim.y*128 zero-padded rows.
// K multiple of 32.  Double-buffered LDS, async global->LDS tile loads.
static constexpr int BM = 256, BN = 128, SA = 40;      // SA: LDS row stride (bf16 elems)
static constexpr int A_ELEMS   = BM * SA;              // 10240
static constexpr int W_ELEMS   = BN * SA;              // 5120
static constexpr int BUF_ELEMS = A_ELEMS + W_ELEMS;    // 15360 (x2 bufs = 61440 B)

__global__ __launch_bounds__(256) void gemm_bf16_wmma(
    const __bf16* __restrict__ A, int lda,
    const __bf16* __restrict__ W, int ldw,
    const float*  __restrict__ bias,
    float*   __restrict__ C,  int ldc,
    __bf16*  __restrict__ Cb, int ldcb,
    int N, int K, int relu)
{
    __shared__ __attribute__((aligned(16))) __bf16 sm[2 * BUF_ELEMS];

    const int tid  = threadIdx.x;
    const int lane = tid & 31;
    const int wave = tid >> 5;
    const int wm   = (wave & 3) * 64;      // wave row offset (4 row-groups)
    const int wn   = (wave >> 2) * 64;     // wave col offset (2 col-groups)
    const int bm   = blockIdx.x * BM;
    const int bn   = blockIdx.y * BN;
    const int half = lane >> 4;            // 0 or 1
    const int l15  = lane & 15;

    const unsigned smbase = lds_lo(sm);

    // per-thread 16B chunk coordinates: A tile 256x32 -> 1024 chunks (4/thread),
    //                                   W tile 128x32 ->  512 chunks (2/thread)
    int ar[4], ac[4], wr[2], wc[2];
#pragma unroll
    for (int t = 0; t < 4; ++t) { int c = tid + t * 256; ar[t] = c >> 2; ac[t] = (c & 3) << 3; }
#pragma unroll
    for (int t = 0; t < 2; ++t) { int c = tid + t * 256; wr[t] = c >> 2; wc[t] = (c & 3) << 3; }

    auto issue = [&](int k0, int buf) {
        const unsigned aoff = smbase + (unsigned)(buf * BUF_ELEMS) * 2u;
        const unsigned woff = aoff + (unsigned)A_ELEMS * 2u;
#pragma unroll
        for (int t = 0; t < 4; ++t)
            async_cp16(aoff + (unsigned)(ar[t] * SA + ac[t]) * 2u,
                       (unsigned long long)(uintptr_t)(A + (size_t)(bm + ar[t]) * lda + k0 + ac[t]));
#pragma unroll
        for (int t = 0; t < 2; ++t)
            async_cp16(woff + (unsigned)(wr[t] * SA + wc[t]) * 2u,
                       (unsigned long long)(uintptr_t)(W + (size_t)(bn + wr[t]) * ldw + k0 + wc[t]));
    };

    v8f acc[4][4];
#pragma unroll
    for (int i = 0; i < 4; ++i)
#pragma unroll
        for (int j = 0; j < 4; ++j) acc[i][j] = {};

    issue(0, 0);
    const int nk = K >> 5;
    for (int it = 0; it < nk; ++it) {
        const int buf = it & 1;
        if (it + 1 < nk) {
            issue((it + 1) << 5, buf ^ 1);
            wait_asynccnt<6>();            // drain current tile (next tile's 6 remain)
        } else {
            wait_asynccnt<0>();
        }
        __syncthreads();

        __bf16* As = sm + buf * BUF_ELEMS;
        __bf16* Ws = As + A_ELEMS;

        // A fragments: 16x32; lane half selects K groups {h*8.., 16+h*8..}
        v16bf afr[4];
#pragma unroll
        for (int i = 0; i < 4; ++i) {
            const __bf16* p = &As[(wm + i * 16 + l15) * SA + half * 8];
            v8bf lo = *(const v8bf*)p;
            v8bf hi = *(const v8bf*)(p + 16);
            afr[i] = __builtin_shufflevector(lo, hi, 0,1,2,3,4,5,6,7,8,9,10,11,12,13,14,15);
        }
        // B fragments: lane = column n; lane half selects K groups {h*8.., 16+h*8..}
        v16bf bfr[4];
#pragma unroll
        for (int j = 0; j < 4; ++j) {
            const __bf16* p = &Ws[(wn + j * 16 + l15) * SA + half * 8];
            v8bf lo = *(const v8bf*)p;
            v8bf hi = *(const v8bf*)(p + 16);
            bfr[j] = __builtin_shufflevector(lo, hi, 0,1,2,3,4,5,6,7,8,9,10,11,12,13,14,15);
        }
#pragma unroll
        for (int i = 0; i < 4; ++i)
#pragma unroll
            for (int j = 0; j < 4; ++j)
                acc[i][j] = __builtin_amdgcn_wmma_f32_16x16x32_bf16(
                    false, afr[i], false, bfr[j], (short)0, acc[i][j], false, false);
        __syncthreads();
    }

    // epilogue: row = bm+wm+i*16+half*8+v, col = bn+wn+j*16+l15
#pragma unroll
    for (int j = 0; j < 4; ++j) {
        int col = bn + wn + j * 16 + l15;
        if (col < N) {
            float bv = bias ? bias[col] : 0.0f;
#pragma unroll
            for (int i = 0; i < 4; ++i) {
                int rbase = bm + wm + i * 16 + half * 8;
#pragma unroll
                for (int v = 0; v < 8; ++v) {
                    float r = acc[i][j][v] + bv;
                    if (relu) r = fmaxf(r, 0.0f);
                    if (C)  C [(size_t)(rbase + v) * ldc  + col] = r;
                    if (Cb) Cb[(size_t)(rbase + v) * ldcb + col] = f2bf(r);
                }
            }
        }
    }
}

// ---------------- packing / prep kernels ----------------
// Wall[l] (1664 x 896), rows >=1600 zero:
// [0..4]=Wi [5..31]=0 [32..63]=Ww [64..463]=Wh [464..479]=0 [480..879]=Wb [880..895]=0
__global__ void pack_lstm_w(const float* __restrict__ Wi, const float* __restrict__ Wh,
                            const float* __restrict__ Wb, const float* __restrict__ Ww,
                            __bf16* __restrict__ Wall)
{
    size_t t = (size_t)blockIdx.x * 256 + threadIdx.x;
    size_t total = (size_t)3 * NP_GATE * KL;
    if (t >= total) return;
    int k = (int)(t % KL);
    size_t nl = t / KL;
    int n = (int)(nl % NP_GATE);
    int l = (int)(nl / NP_GATE);
    float v = 0.0f;
    if (n < GATE) {
        size_t rb = (size_t)l * GATE + n;
        if (k < 5)                    v = Wi[rb * 5   + k];
        else if (k >= 32 && k < 64)   v = Ww[rb * 32  + (k - 32)];
        else if (k >= 64 && k < 464)  v = Wh[rb * HID + (k - 64)];
        else if (k >= 480 && k < 880) v = Wb[rb * HID + (k - 480)];
    }
    Wall[t] = f2bf(v);
}

__global__ void pack_pad(const float* __restrict__ src, __bf16* __restrict__ dst,
                         int N, int K, int Np, int Kp)
{
    size_t t = (size_t)blockIdx.x * 256 + threadIdx.x;
    if (t >= (size_t)Np * Kp) return;
    int k = (int)(t % Kp);
    int n = (int)(t / Kp);
    dst[t] = f2bf((n < N && k < K) ? src[(size_t)n * K + k] : 0.0f);
}

// Aall cols 0..63 = [x(4), r(1), 0.., d(32)]; zero cols 464..895; zero hat pad 1200..1215
__global__ void prep_acts(const float* __restrict__ x, const float* __restrict__ r,
                          const float* __restrict__ d,
                          __bf16* __restrict__ Aall, __bf16* __restrict__ hat)
{
    int t = blockIdx.x * 256 + threadIdx.x;
    int b = t >> 9;
    int j = t & 511;
    if (b >= BATCH) return;
    if (j < 64) {
        float v = 0.0f;
        if (j < 4)        v = x[(size_t)b * 4 + j];
        else if (j == 4)  v = r[b];
        else if (j >= 32) v = d[(size_t)b * 32 + (j - 32)];
        Aall[(size_t)b * KL + j] = f2bf(v);
    } else if (j < 496) {
        Aall[(size_t)b * KL + 464 + (j - 64)] = f2bf(0.0f);
    } else {
        hat[(size_t)b * KH + 1200 + (j - 496)] = f2bf(0.0f);
    }
}

__global__ void load_prevh(const float* __restrict__ prev_h, __bf16* __restrict__ Aall, int l)
{
    int t = blockIdx.x * 256 + threadIdx.x;
    if (t >= BATCH * HID) return;
    int b = t / HID, k = t % HID;
    Aall[(size_t)b * KL + 64 + k] = f2bf(prev_h[((size_t)l * BATCH + b) * HID + k]);
}

// gates + state update; h feeds next layer ("below") and hat
__global__ void gates(const float* __restrict__ z, const float* __restrict__ prev_c,
                      float* __restrict__ out, __bf16* __restrict__ Aall,
                      __bf16* __restrict__ hat, int l)
{
    int t = blockIdx.x * 256 + threadIdx.x;
    if (t >= BATCH * HID) return;
    int b = t / HID, j = t % HID;
    const float* zr = z + (size_t)b * GATE;
    float ig = sigf(zr[j]);
    float fg = sigf(zr[HID + j]);
    float og = sigf(zr[2 * HID + j]);
    float gg = tanhf(zr[3 * HID + j]);
    float cp = prev_c[((size_t)l * BATCH + b) * HID + j];
    float c  = fg * cp + ig * gg;
    float h  = og * tanhf(c);
    float* nh = out + (size_t)BATCH * 123;
    float* nc = nh + (size_t)3 * BATCH * HID;
    nh[((size_t)l * BATCH + b) * HID + j] = h;
    nc[((size_t)l * BATCH + b) * HID + j] = c;
    Aall[(size_t)b * KL + 480 + j] = f2bf(h);
    hat[(size_t)b * KH + (size_t)l * HID + j] = f2bf(h);
}

// final head: e/softmax/mu/rho/sigma + r1,r2 from f3 (32-wide dots)
__global__ void head(const float* __restrict__ hy, const __bf16* __restrict__ f3,
                     const float* __restrict__ Wr1, const float* __restrict__ br1,
                     const float* __restrict__ Wr2, const float* __restrict__ br2,
                     float* __restrict__ out)
{
    int b = blockIdx.x * 256 + threadIdx.x;
    if (b >= BATCH) return;
    const float* h = hy + (size_t)b * 128;
    float* o = out + (size_t)b * 123;
    o[0] = 1.0f / (1.0f + expf(h[0]));               // sigmoid(-hy0)
    float mx = h[1];
    for (int k = 2; k <= 20; ++k) mx = fmaxf(mx, h[k]);
    float e[20], s = 0.0f;
    for (int k = 0; k < 20; ++k) { e[k] = expf(h[1 + k] - mx); s += e[k]; }
    float inv = 1.0f / s;
    for (int k = 0; k < 20; ++k) o[1 + k]   = e[k] * inv;
    for (int k = 0; k < 20; ++k) o[21 + k]  = h[21 + k];
    for (int k = 0; k < 20; ++k) o[41 + k]  = h[41 + k];
    for (int k = 0; k < 20; ++k) o[61 + k]  = tanhf(h[61 + k]);
    for (int k = 0; k < 20; ++k) o[81 + k]  = expf(h[81 + k]) + 0.001f;
    for (int k = 0; k < 20; ++k) o[101 + k] = expf(h[101 + k]) + 0.001f;
    const __bf16* f = f3 + (size_t)b * 32;
    float a1 = br1[0], a2 = br2[0];
    for (int k = 0; k < 32; ++k) {
        float fv = bf2f(f[k]);
        a1 += fv * Wr1[k];
        a2 += fv * Wr2[k];
    }
    o[121] = tanhf(a1);
    o[122] = sigf(a2);
}

// ---------------- launch ----------------
extern "C" void kernel_launch(void* const* d_in, const int* in_sizes, int n_in,
                              void* d_out, int out_size, void* d_ws, size_t ws_size,
                              hipStream_t stream) {
    (void)in_sizes; (void)n_in; (void)out_size; (void)ws_size;
    const float* x      = (const float*)d_in[0];
    const float* r      = (const float*)d_in[1];
    const float* dd     = (const float*)d_in[2];
    const float* prev_h = (const float*)d_in[3];
    const float* prev_c = (const float*)d_in[4];
    const float* Wi     = (const float*)d_in[5];
    const float* Wh     = (const float*)d_in[6];
    const float* Wb     = (const float*)d_in[7];
    const float* Ww     = (const float*)d_in[8];
    const float* bi     = (const float*)d_in[9];
    const float* Wy     = (const float*)d_in[10];
    const float* by     = (const float*)d_in[11];
    const float* W1     = (const float*)d_in[12];
    const float* b1     = (const float*)d_in[13];
    const float* W2     = (const float*)d_in[14];
    const float* b2     = (const float*)d_in[15];
    const float* W3     = (const float*)d_in[16];
    const float* b3     = (const float*)d_in[17];
    const float* Wr1    = (const float*)d_in[18];
    const float* br1    = (const float*)d_in[19];
    const float* Wr2    = (const float*)d_in[20];
    const float* br2    = (const float*)d_in[21];
    float* out = (float*)d_out;

    uintptr_t wp = (uintptr_t)d_ws;
    auto alloc = [&](size_t bytes) -> void* {
        uintptr_t p = (wp + 255) & ~(uintptr_t)255;
        wp = p + bytes;
        return (void*)p;
    };
    __bf16* Wall = (__bf16*)alloc((size_t)3 * NP_GATE * KL * 2);
    __bf16* WyP  = (__bf16*)alloc((size_t)128 * KH * 2);
    __bf16* W1P  = (__bf16*)alloc((size_t)256 * KH * 2);
    __bf16* W2P  = (__bf16*)alloc((size_t)128 * 256 * 2);
    __bf16* W3P  = (__bf16*)alloc((size_t)128 * 64  * 2);
    __bf16* Aall = (__bf16*)alloc((size_t)BATCH * KL * 2);
    __bf16* hat  = (__bf16*)alloc((size_t)BATCH * KH * 2);
    float*  z    = (float*) alloc((size_t)BATCH * GATE * 4);
    float*  hyb  = (float*) alloc((size_t)BATCH * 128 * 4);
    __bf16* f1b  = (__bf16*)alloc((size_t)BATCH * 256 * 2);
    __bf16* f2b  = (__bf16*)alloc((size_t)BATCH * 64  * 2);
    __bf16* f3b  = (__bf16*)alloc((size_t)BATCH * 32  * 2);

    // weight packing (fp32 -> N/K zero-padded bf16)
    {
        size_t tw = (size_t)3 * NP_GATE * KL;
        pack_lstm_w<<<dim3((unsigned)((tw + 255) / 256)), dim3(256), 0, stream>>>(Wi, Wh, Wb, Ww, Wall);
        pack_pad<<<dim3((unsigned)(((size_t)128 * KH  + 255) / 256)), dim3(256), 0, stream>>>(Wy, WyP, YD,  1200, 128, KH);
        pack_pad<<<dim3((unsigned)(((size_t)256 * KH  + 255) / 256)), dim3(256), 0, stream>>>(W1, W1P, 256, 1200, 256, KH);
        pack_pad<<<dim3((unsigned)(((size_t)128 * 256 + 255) / 256)), dim3(256), 0, stream>>>(W2, W2P, 64,  256,  128, 256);
        pack_pad<<<dim3((unsigned)(((size_t)128 * 64  + 255) / 256)), dim3(256), 0, stream>>>(W3, W3P, 32,  64,   128, 64);
    }
    prep_acts<<<dim3(BATCH * 512 / 256), dim3(256), 0, stream>>>(x, r, dd, Aall, hat);

    const dim3 blk(256);
    for (int l = 0; l < 3; ++l) {
        load_prevh<<<dim3(BATCH * HID / 256), blk, 0, stream>>>(prev_h, Aall, l);
        dim3 g(BATCH / BM, (GATE + BN - 1) / BN);   // 32 x 13
        gemm_bf16_wmma<<<g, blk, 0, stream>>>(Aall, KL, Wall + (size_t)l * NP_GATE * KL, KL,
                                              bi + (size_t)l * GATE,
                                              z, GATE, (__bf16*)nullptr, 0,
                                              GATE, KL, 0);
        gates<<<dim3(BATCH * HID / 256), blk, 0, stream>>>(z, prev_c, out, Aall, hat, l);
    }

    gemm_bf16_wmma<<<dim3(BATCH / BM, 1), blk, 0, stream>>>(hat, KH, WyP, KH, by,
                                                            hyb, 128, (__bf16*)nullptr, 0,
                                                            YD, KH, 0);
    gemm_bf16_wmma<<<dim3(BATCH / BM, 2), blk, 0, stream>>>(hat, KH, W1P, KH, b1,
                                                            (float*)nullptr, 0, f1b, 256,
                                                            256, KH, 1);
    gemm_bf16_wmma<<<dim3(BATCH / BM, 1), blk, 0, stream>>>(f1b, 256, W2P, 256, b2,
                                                            (float*)nullptr, 0, f2b, 64,
                                                            64, 256, 1);
    gemm_bf16_wmma<<<dim3(BATCH / BM, 1), blk, 0, stream>>>(f2b, 64, W3P, 64, b3,
                                                            (float*)nullptr, 0, f3b, 32,
                                                            32, 64, 1);
    head<<<dim3(BATCH / 256), blk, 0, stream>>>(hyb, f3b, Wr1, br1, Wr2, br2, out);
}
